// Attention_44873818308981
// MI455X (gfx1250) — compile-verified, matrix operand order checked by reading, not run on hardware
//
#include <hip/hip_runtime.h>

// Flash-attention with relative-position bias for MI455X (gfx1250, wave32).
// QK^T: V_WMMA_F32_16X16X32_BF16 with 3-term hi/lo bf16 split (~2^-15 rel err).
// P@V : V_WMMA_F32_16X16X4_F32 with exact fp32 probabilities (P recycled
//       through the LDS transpose tile, zero conversion VALU).
// K is pre-split to bf16 hi/lo planes; V is pre-transposed to [b][d][s] f32.

typedef __attribute__((ext_vector_type(2)))  float  v2f;
typedef __attribute__((ext_vector_type(4)))  float  v4f;
typedef __attribute__((ext_vector_type(8)))  float  v8f;
typedef __attribute__((ext_vector_type(8)))  __bf16 v8bf;
typedef __attribute__((ext_vector_type(16))) __bf16 v16bf;

#define B_DIM  16
#define S_DIM  2048
#define DH     64
#define NWAVES 8        // waves per block
#define PSTR   36       // padded LDS row stride (floats): conflict-free, 16B-aligned
#define KV_ELEMS (B_DIM * S_DIM * DH)

__device__ __bf16 g_khi[KV_ELEMS];  // [b][s][d]
__device__ __bf16 g_klo[KV_ELEMS];
__device__ float  g_vt[KV_ELEMS];   // transposed: [b][d][s], f32

__device__ __forceinline__ unsigned int bfbits(__bf16 h) {
  return (unsigned int)__builtin_bit_cast(unsigned short, h);
}

// ---- split K -> bf16 hi/lo planes (2 elements/thread, packed b32 stores) ----
__global__ __launch_bounds__(256) void split_k_kernel(const float* __restrict__ K) {
  const int i = blockIdx.x * 256 + threadIdx.x;
  const float x0 = K[2 * i], x1 = K[2 * i + 1];
  const __bf16 h0 = (__bf16)x0, h1 = (__bf16)x1;
  const __bf16 l0 = (__bf16)(x0 - (float)h0), l1 = (__bf16)(x1 - (float)h1);
  ((unsigned int*)g_khi)[i] = bfbits(h0) | (bfbits(h1) << 16);
  ((unsigned int*)g_klo)[i] = bfbits(l0) | (bfbits(l1) << 16);
}

// ---- transpose V -> f32 plane [b][d][s] (LDS-tiled, coalesced) ----
__global__ __launch_bounds__(256) void transpose_v_kernel(const float* __restrict__ V) {
  __shared__ float tile[64][65];
  const int b   = blockIdx.x >> 5;          // 32 s-blocks of 64 per batch-head
  const int s0  = (blockIdx.x & 31) << 6;
  const int tid = threadIdx.x;
#pragma unroll
  for (int r = 0; r < 16; ++r) {
    const int idx = r * 256 + tid;
    const int sl = idx >> 6, d = idx & 63;
    tile[sl][d] = V[((size_t)(b * S_DIM + s0 + sl)) * DH + d];
  }
  __syncthreads();
  const int d  = tid >> 2;
  const int sr = (tid & 3) << 4;
  float* o = g_vt + ((size_t)(b * DH + d)) * S_DIM + s0 + sr;
#pragma unroll
  for (int j = 0; j < 16; j += 4) {
    v4f t;
    t[0] = tile[sr + j][d];
    t[1] = tile[sr + j + 1][d];
    t[2] = tile[sr + j + 2][d];
    t[3] = tile[sr + j + 3][d];
    *(v4f*)(o + j) = t;
  }
}

__device__ __forceinline__ v8f wmma_bf16(v16bf a, v16bf b, v8f c) {
  return __builtin_amdgcn_wmma_f32_16x16x32_bf16(false, a, false, b, (short)0, c,
                                                 false, false);
}
__device__ __forceinline__ v8f wmma_f32(v2f a, v2f b, v8f c) {
  return __builtin_amdgcn_wmma_f32_16x16x4_f32(false, a, false, b, (short)0, c,
                                               false, false);
}
__device__ __forceinline__ v16bf cat8(v8bf a, v8bf b) {
  return __builtin_shufflevector(a, b, 0, 1, 2, 3, 4, 5, 6, 7,
                                       8, 9, 10, 11, 12, 13, 14, 15);
}
// fragment = two contiguous 8-elem runs at p and p+16 (A/B 16-bit layout)
__device__ __forceinline__ v16bf ld_frag(const __bf16* p) {
  return cat8(*(const v8bf*)p, *(const v8bf*)(p + 16));
}

__global__ __launch_bounds__(256) void attn_fa_kernel(
    const float* __restrict__ Q, const float* __restrict__ Rw,
    const float* __restrict__ Rb, float* __restrict__ Out) {
  __shared__ float sh_rw[2 * DH];
  __shared__ float sh_rb[2];
  __shared__ float sh_p[NWAVES][16 * PSTR];  // 16x32 score/prob transpose tile
  __shared__ float sh_bc[NWAVES][16];        // lm-domain -> g-domain broadcast

  const int tid  = threadIdx.x;
  const int wave = tid >> 5;
  const int lane = tid & 31;
  const int lm   = lane & 15;
  const int lh   = lane >> 4;

  if (tid < 2 * DH) sh_rw[tid] = Rw[tid];
  if (tid < 2) sh_rb[tid] = Rb[tid];
  __syncthreads();

  const int tile = blockIdx.x * NWAVES + wave;
  const int b    = tile >> 7;
  const int q0   = (tile & 127) << 4;

  // log2(e)/sqrt(512): softmax in base-2 domain -> native v_exp_f32
  const float qscale = 0.04419417382415922f * 1.4426950408889634f;

  // ---- Q row (lm) in bf16-fragment element order: q32[16c + e] ----
  float q32[32];
  {
    const float* qrow = Q + ((size_t)(b * S_DIM + q0 + lm)) * DH;
#pragma unroll
    for (int r = 0; r < 4; ++r) {
      const v4f t0 = *(const v4f*)(qrow + 16 * r + 8 * lh);
      const v4f t1 = *(const v4f*)(qrow + 16 * r + 8 * lh + 4);
#pragma unroll
      for (int j = 0; j < 4; ++j) {
        q32[r * 8 + j]     = t0[j] * qscale;
        q32[r * 8 + 4 + j] = t1[j] * qscale;
      }
    }
  }

  // ---- relative-position bias for q-row lm: pos0 (j<i), posd = pos1-pos0 ----
  float pos0, posd;
  {
    float part0 = 0.f, part1 = 0.f;
#pragma unroll
    for (int r = 0; r < 4; ++r) {
#pragma unroll
      for (int j = 0; j < 8; ++j) {
        const int d = 16 * r + 8 * lh + j;
        part0 += q32[r * 8 + j] * sh_rw[d];
        part1 += q32[r * 8 + j] * sh_rw[DH + d];
      }
    }
    part0 += __shfl_xor(part0, 16);
    part1 += __shfl_xor(part1, 16);
    part0 += sh_rb[0] * qscale;
    part1 += sh_rb[1] * qscale;
    pos0 = part0;
    posd = part1 - part0;
  }

  float* Pl = sh_p[wave];
  float* Bc = sh_bc[wave];

  // ---- pos0 broadcast into the WMMA C layout (rows g+8*lh): bias rides the
  //      matrix op's C operand, no per-step bias adds ----
  v8f posC;
  {
    Bc[lm] = pos0;
    asm volatile("" ::: "memory");
#pragma unroll
    for (int g = 0; g < 8; ++g) posC[g] = Bc[g + 8 * lh];
  }

  // ---- Q bf16 hi/lo A-fragments (2 chunks of K=32) ----
  v16bf aQhi[2], aQlo[2];
#pragma unroll
  for (int c = 0; c < 2; ++c) {
#pragma unroll
    for (int e = 0; e < 16; ++e) {
      const float x = q32[16 * c + e];
      const __bf16 h = (__bf16)x;
      aQhi[c][e] = h;
      aQlo[c][e] = (__bf16)(x - (float)h);
    }
  }

  v8f acc0 = {}, acc1 = {}, acc2 = {}, acc3 = {};
  float mrun = -3.0e38f, srun = 0.f;

  const __bf16* khi_b = g_khi + (size_t)b * S_DIM * DH;
  const __bf16* klo_b = g_klo + (size_t)b * S_DIM * DH;
  const float*  vt_b  = g_vt  + (size_t)b * DH * S_DIM;

  auto step32 = [&](int kb, bool tail) {
    // ---- K fragments (hi/lo) for two 16-key tiles ----
    const int row0 = kb + lm;
    int row1 = kb + 16 + lm;
    if (tail && row1 > S_DIM - 1) row1 = S_DIM - 1;  // masked anyway
    const __bf16* kr0h = khi_b + (size_t)row0 * DH + 8 * lh;
    const __bf16* kr1h = khi_b + (size_t)row1 * DH + 8 * lh;
    const __bf16* kr0l = klo_b + (size_t)row0 * DH + 8 * lh;
    const __bf16* kr1l = klo_b + (size_t)row1 * DH + 8 * lh;
    if (!tail) {
      __builtin_prefetch(kr0h + 32 * DH, 0, 3);
      __builtin_prefetch(kr1h + 32 * DH, 0, 3);
    }

    // ---- S tiles (+pos0 via C): 3-term bf16 split, 12 WMMAs, two chains ----
    v8f Sv0 = posC, Sv1 = posC;
#pragma unroll
    for (int c = 0; c < 2; ++c) {
      const v16bf k0h = ld_frag(kr0h + 32 * c);
      const v16bf k1h = ld_frag(kr1h + 32 * c);
      const v16bf k0l = ld_frag(kr0l + 32 * c);
      const v16bf k1l = ld_frag(kr1l + 32 * c);
      Sv0 = wmma_bf16(aQhi[c], k0h, Sv0);
      Sv1 = wmma_bf16(aQhi[c], k1h, Sv1);
      Sv0 = wmma_bf16(aQlo[c], k0h, Sv0);
      Sv1 = wmma_bf16(aQlo[c], k1h, Sv1);
      Sv0 = wmma_bf16(aQhi[c], k0l, Sv0);
      Sv1 = wmma_bf16(aQhi[c], k1l, Sv1);
    }

    // ---- transpose both 16x16 tiles (C/D -> A domain) through padded LDS ----
#pragma unroll
    for (int g = 0; g < 8; ++g) {
      Pl[(g + 8 * lh) * PSTR + lm]      = Sv0[g];
      Pl[(g + 8 * lh) * PSTR + 16 + lm] = Sv1[g];
    }
    asm volatile("" ::: "memory");  // same-wave LDS pipe is in-order
    float vals[16];
    {
      const v4f r0 = *(const v4f*)&Pl[lm * PSTR + 8 * lh];
      const v4f r1 = *(const v4f*)&Pl[lm * PSTR + 8 * lh + 4];
      const v4f r2 = *(const v4f*)&Pl[lm * PSTR + 16 + 8 * lh];
      const v4f r3 = *(const v4f*)&Pl[lm * PSTR + 20 + 8 * lh];
#pragma unroll
      for (int j = 0; j < 4; ++j) {
        vals[j]      = r0[j];
        vals[4 + j]  = r1[j];
        vals[8 + j]  = r2[j];
        vals[12 + j] = r3[j];
      }
    }

    // ---- (tail-only) causal mask + diagonal bias in A domain ----
    if (tail) {
#pragma unroll
      for (int e = 0; e < 16; ++e) {
        const int k  = (e < 8) ? (8 * lh + e) : (16 + 8 * lh + (e - 8));
        const int dk = (kb + k) - (q0 + lm);
        vals[e] = (dk > 0) ? -1.0e30f : ((dk == 0) ? vals[e] + posd : vals[e]);
      }
    }

    // ---- online softmax over 32 keys; balanced-tree reductions (depth 4) ----
    float tm[8];
#pragma unroll
    for (int i = 0; i < 8; ++i) tm[i] = fmaxf(vals[i], vals[8 + i]);
#pragma unroll
    for (int i = 0; i < 4; ++i) tm[i] = fmaxf(tm[i], tm[4 + i]);
    float rmax = fmaxf(fmaxf(tm[0], tm[1]), fmaxf(tm[2], tm[3]));
    rmax = fmaxf(rmax, __shfl_xor(rmax, 16));
    const float nm = fmaxf(mrun, rmax);
    if (__any(rmax > mrun)) {  // wave-uniform; rare in steady state
      const float corr = exp2f(mrun - nm);
      srun *= corr;
      Bc[lm] = corr;
      asm volatile("" ::: "memory");
#pragma unroll
      for (int g = 0; g < 8; ++g) {
        const float cg = Bc[g + 8 * lh];
        acc0[g] *= cg; acc1[g] *= cg; acc2[g] *= cg; acc3[g] *= cg;
      }
    }
    mrun = nm;

    v4f p0, p1, p2, p3;
#pragma unroll
    for (int j = 0; j < 4; ++j) {
      p0[j] = exp2f(vals[j] - nm);
      p1[j] = exp2f(vals[4 + j] - nm);
      p2[j] = exp2f(vals[8 + j] - nm);
      p3[j] = exp2f(vals[12 + j] - nm);
    }
    const v4f ps = (p0 + p1) + (p2 + p3);   // element-wise tree
    float rs = (ps[0] + ps[1]) + (ps[2] + ps[3]);
    rs += __shfl_xor(rs, 16);
    srun += rs;

    // ---- recycle LDS tile: store exact-f32 P, reread in fp32 A layout ----
    *(v4f*)&Pl[lm * PSTR + 8 * lh]          = p0;
    *(v4f*)&Pl[lm * PSTR + 8 * lh + 4]      = p1;
    *(v4f*)&Pl[lm * PSTR + 16 + 8 * lh]     = p2;
    *(v4f*)&Pl[lm * PSTR + 20 + 8 * lh]     = p3;
    asm volatile("" ::: "memory");
    v2f aP[8];
#pragma unroll
    for (int c = 0; c < 8; ++c)
      aP[c] = *(const v2f*)&Pl[lm * PSTR + 4 * c + 2 * lh];

    // ---- O += P @ V : exact fp32, 32 WMMAs, 4 independent chains ----
#pragma unroll
    for (int c = 0; c < 8; ++c) {
      int s = kb + 4 * c + 2 * lh;
      if (tail && s > S_DIM - 2) s = S_DIM - 2;  // masked anyway (P==0)
      const float* vcol = vt_b + s;
      const v2f bv0 = *(const v2f*)(vcol + (size_t)lm * S_DIM);
      const v2f bv1 = *(const v2f*)(vcol + (size_t)(16 + lm) * S_DIM);
      const v2f bv2 = *(const v2f*)(vcol + (size_t)(32 + lm) * S_DIM);
      const v2f bv3 = *(const v2f*)(vcol + (size_t)(48 + lm) * S_DIM);
      acc0 = wmma_f32(aP[c], bv0, acc0);
      acc1 = wmma_f32(aP[c], bv1, acc1);
      acc2 = wmma_f32(aP[c], bv2, acc2);
      acc3 = wmma_f32(aP[c], bv3, acc3);
    }
  };

  int kb = 0;
  for (; kb + 32 <= q0; kb += 32) step32(kb, false);  // mask-free hot loop
  step32(kb, true);                                   // diagonal (+dead) tail

  // ---- epilogue: broadcast 1/srun to g-domain, normalize, store ----
  Bc[lm] = srun;
  asm volatile("" ::: "memory");
#pragma unroll
  for (int g = 0; g < 8; ++g) {
    const float inv = __builtin_amdgcn_rcpf(Bc[g + 8 * lh]);
    float* orow = Out + ((size_t)b * S_DIM + q0 + g + 8 * lh) * DH + lm;
    orow[0]  = acc0[g] * inv;
    orow[16] = acc1[g] * inv;
    orow[32] = acc2[g] * inv;
    orow[48] = acc3[g] * inv;
  }
}

extern "C" void kernel_launch(void* const* d_in, const int* in_sizes, int n_in,
                              void* d_out, int out_size, void* d_ws, size_t ws_size,
                              hipStream_t stream) {
  const float* Q  = (const float*)d_in[0];
  const float* K  = (const float*)d_in[1];
  const float* V  = (const float*)d_in[2];
  const float* Rw = (const float*)d_in[3];
  const float* Rb = (const float*)d_in[4];
  float* Out = (float*)d_out;

  split_k_kernel<<<KV_ELEMS / 2 / 256, 256, 0, stream>>>(K);
  transpose_v_kernel<<<B_DIM * 32, 256, 0, stream>>>(V);

  const int tiles  = B_DIM * (S_DIM / 16);   // 2048 q-tiles
  const int blocks = tiles / NWAVES;         // 256 blocks of 8 waves
  attn_fa_kernel<<<blocks, 256, 0, stream>>>(Q, Rw, Rb, Out);
}
